// GPT2Block_7516192768319
// MI455X (gfx1250) — matrix operand order repaired and checked by reading, not compiled, not run on hardware
//
#include <hip/hip_runtime.h>
#include <math.h>

// GPT-2 block: B=4, S=2048, D=1024, H=16, HD=64
#define B_  4
#define S_  2048
#define D_  1024
#define H_  16
#define HD_ 64
#define AW  8   // waves per attention block

typedef __bf16 bf16;
typedef __attribute__((ext_vector_type(16))) __bf16 v16bf;
typedef __attribute__((ext_vector_type(8)))  __bf16 v8bf;
typedef __attribute__((ext_vector_type(8)))  float  v8f;
typedef __attribute__((ext_vector_type(4)))  unsigned int v4u;
typedef __attribute__((ext_vector_type(8)))  int v8i;
typedef __attribute__((ext_vector_type(4)))  int v4i;

// ---------------------------------------------------------------------------
// WMMA helpers (CDNA5 16x16x32 bf16, f32 accumulate)
// ---------------------------------------------------------------------------
__device__ __forceinline__ v8f zero_v8f() {
    v8f z;
#pragma unroll
    for (int i = 0; i < 8; ++i) z[i] = 0.0f;
    return z;
}

__device__ __forceinline__ v16bf combine16(v8bf lo, v8bf hi) {
    v16bf f;
#pragma unroll
    for (int i = 0; i < 8; ++i) { f[i] = lo[i]; f[i + 8] = hi[i]; }
    return f;
}

// A-operand 16x32 bf16 tile fragment. Lane L holds row (L%16); the 16 values
// per lane are K = k0 + 8*half + [0..7] and K = k0 + 16 + 8*half + [0..7].
__device__ __forceinline__ v16bf load_a_frag(const bf16* rowptr, int k0, int half) {
    v8bf lo = *(const v8bf*)(rowptr + k0 + 8 * half);
    v8bf hi = *(const v8bf*)(rowptr + k0 + 16 + 8 * half);
    return combine16(lo, hi);
}

// B-operand 32x16 bf16 tile fragment from a B^T ([N x K] row-major) source.
// Lane L holds column N=(L%16); values are K = k0 + 8*half + [0..7] and K = k0 + 16 + 8*half + [0..7].
__device__ __forceinline__ v16bf load_b_frag(const bf16* rowptr, int k0, int half) {
    const bf16* p = rowptr + k0 + 8 * half;
    v8bf lo = *(const v8bf*)p;
    v8bf hi = *(const v8bf*)(p + 16);
    return combine16(lo, hi);
}

__device__ __forceinline__ v8f wmma_bf16(v16bf a, v16bf b, v8f c) {
    // (neg_a, A, neg_b, B, c_mod, C, reuse_a, reuse_b)
    return __builtin_amdgcn_wmma_f32_16x16x32_bf16(false, a, false, b,
                                                   (short)0, c, false, false);
}

// ---------------------------------------------------------------------------
// TDM: issue a 2D tensor tile load (64 rows x 32 bf16) global -> LDS.
// Descriptor built per CDNA5 D# group0/group1 layout (CDNA5 ISA).
// Tracked by TENSORcnt; caller pairs with s_wait_tensorcnt + barrier.
// ---------------------------------------------------------------------------
__device__ __forceinline__ void tdm_load_tile64x32(unsigned lds_off,
                                                   const bf16* gaddr,
                                                   int tdim0, int tdim1,
                                                   int stride0) {
    const unsigned long long ga = (unsigned long long)(uintptr_t)gaddr;
    v4u g0;
    g0[0] = 1u;                                   // count=1, no gather
    g0[1] = lds_off;                              // lds_addr (bytes)
    g0[2] = (unsigned)(ga & 0xFFFFFFFFu);         // global_addr[31:0]
    g0[3] = (unsigned)((ga >> 32) & 0x01FFFFFFu)  // global_addr[56:32]
          | (2u << 30);                           // type=2 ("image")
    v8i g1;
    g1[0] = (int)(1u << 16);                      // wg_mask=0, data_size=2B
    g1[1] = (int)(((unsigned)tdim0 & 0xFFFFu) << 16);            // tensor_dim0 lo
    g1[2] = (int)((((unsigned)tdim0 >> 16) & 0xFFFFu)
          | (((unsigned)tdim1 & 0xFFFFu) << 16));                // td0 hi | td1 lo
    g1[3] = (int)((((unsigned)tdim1 >> 16) & 0xFFFFu)
          | (32u << 16));                                        // tile_dim0 = 32
    g1[4] = 64;                                   // tile_dim1 = 64 rows
    g1[5] = stride0;                              // tensor_dim0_stride lo32
    g1[6] = 0;
    g1[7] = 0;
    v4i z4 = {0, 0, 0, 0};
#if __clang_major__ >= 23
    v8i z8 = {0, 0, 0, 0, 0, 0, 0, 0};
    __builtin_amdgcn_tensor_load_to_lds(g0, g1, z4, z4, z8, 0);
#else
    __builtin_amdgcn_tensor_load_to_lds(g0, g1, z4, z4, 0);
#endif
}

// ---------------------------------------------------------------------------
// Weight convert + transpose: fp32 [R x C] row-major -> bf16 [C x R] row-major
// ---------------------------------------------------------------------------
__global__ __launch_bounds__(256)
void wconv_kernel(const float* __restrict__ in, bf16* __restrict__ out,
                  int R, int C) {
    __shared__ float t[32][33];
    const int tx = threadIdx.x, ty = threadIdx.y;
    const int c = blockIdx.x * 32 + tx;
    const int r0 = blockIdx.y * 32;
#pragma unroll
    for (int i = 0; i < 4; ++i)
        t[ty + i * 8][tx] = in[(size_t)(r0 + ty + i * 8) * C + c];
    __syncthreads();
    const int r = r0 + tx;
    const int c0 = blockIdx.x * 32;
#pragma unroll
    for (int i = 0; i < 4; ++i)
        out[(size_t)(c0 + ty + i * 8) * R + r] = (bf16)t[tx][ty + i * 8];
}

// ---------------------------------------------------------------------------
// LayerNorm: fp32 row [D] -> bf16 row
// ---------------------------------------------------------------------------
__global__ __launch_bounds__(256)
void ln_kernel(const float* __restrict__ x, const float* __restrict__ g,
               const float* __restrict__ b, bf16* __restrict__ out) {
    __shared__ float red[256];
    const int row = blockIdx.x;
    const float* xr = x + (size_t)row * D_;
    float v[4];
    float s = 0.0f;
#pragma unroll
    for (int i = 0; i < 4; ++i) { v[i] = xr[threadIdx.x + i * 256]; s += v[i]; }
    red[threadIdx.x] = s;
    __syncthreads();
    for (int off = 128; off > 0; off >>= 1) {
        if (threadIdx.x < off) red[threadIdx.x] += red[threadIdx.x + off];
        __syncthreads();
    }
    const float mu = red[0] * (1.0f / D_);
    __syncthreads();
    float s2 = 0.0f;
#pragma unroll
    for (int i = 0; i < 4; ++i) { float d = v[i] - mu; s2 += d * d; }
    red[threadIdx.x] = s2;
    __syncthreads();
    for (int off = 128; off > 0; off >>= 1) {
        if (threadIdx.x < off) red[threadIdx.x] += red[threadIdx.x + off];
        __syncthreads();
    }
    const float rstd = rsqrtf(red[0] * (1.0f / D_) + 1e-5f);
#pragma unroll
    for (int i = 0; i < 4; ++i) {
        const int c = threadIdx.x + i * 256;
        out[(size_t)row * D_ + c] = (bf16)((v[i] - mu) * rstd * g[c] + b[c]);
    }
}

// ---------------------------------------------------------------------------
// bf16 WMMA GEMM: C[M,N] = act(A[M,K] @ Bt[N,K]^T + bias) (+res)
// Block = 256 threads = 8 waves; block tile 128x64; wave tile 32x32.
// The 64-row B^T tile for each K-step is staged in LDS by the Tensor Data
// Mover (double-buffered, TENSORcnt-ordered); A streams from global.
// ---------------------------------------------------------------------------
template <int GELU, int OUTBF16, int RES>
__global__ __launch_bounds__(256)
void gemm_kernel(const bf16* __restrict__ A, int lda,
                 const bf16* __restrict__ Bt, int ldb,
                 const float* __restrict__ bias,
                 const float* __restrict__ res,
                 float* __restrict__ Cf, bf16* __restrict__ Cb,
                 int M, int N, int K) {
    __shared__ __align__(16) bf16 bsh[2][64 * 32];
    const int wave = threadIdx.x >> 5;
    const int lane = threadIdx.x & 31;
    const int half = lane >> 4;
    const int lr   = lane & 15;
    const int wm = wave & 3, wn = wave >> 2;
    const int m0 = blockIdx.y * 128 + wm * 32;
    const int n0blk = blockIdx.x * 64;

    v8f acc[2][2];
#pragma unroll
    for (int i = 0; i < 2; ++i)
#pragma unroll
        for (int j = 0; j < 2; ++j) acc[i][j] = zero_v8f();

    const bf16* arow0 = A + (size_t)(m0 + lr) * lda;
    const bf16* arow1 = A + (size_t)(m0 + 16 + lr) * lda;
    const bf16* gB = Bt + (size_t)n0blk * ldb;   // 64-row B^T panel for block

    const unsigned lds0 = (unsigned)(uintptr_t)&bsh[0][0];
    const unsigned lds1 = (unsigned)(uintptr_t)&bsh[1][0];

    // prologue: stage first K-chunk
    if (threadIdx.x < 32) tdm_load_tile64x32(lds0, gB, K, N, ldb);

    int cur = 0;
    for (int k0 = 0; k0 < K; k0 += 32) {
        if (k0 + 32 < K) {
            if (threadIdx.x < 32)
                tdm_load_tile64x32(cur ? lds0 : lds1, gB + k0 + 32, K, N, ldb);
            __builtin_amdgcn_s_wait_tensorcnt(1);   // current chunk landed
        } else {
            __builtin_amdgcn_s_wait_tensorcnt(0);
        }
        __syncthreads();

        __builtin_prefetch(arow0 + k0 + 64, 0, 0);  // global_prefetch_b8

        const bf16* brow0 = &bsh[cur][(wn * 32 + lr) * 32];
        const bf16* brow1 = &bsh[cur][(wn * 32 + 16 + lr) * 32];
        v16bf a0 = load_a_frag(arow0, k0, half);
        v16bf a1 = load_a_frag(arow1, k0, half);
        v16bf b0 = load_b_frag(brow0, 0, half);
        v16bf b1 = load_b_frag(brow1, 0, half);
        acc[0][0] = wmma_bf16(a0, b0, acc[0][0]);
        acc[0][1] = wmma_bf16(a0, b1, acc[0][1]);
        acc[1][0] = wmma_bf16(a1, b0, acc[1][0]);
        acc[1][1] = wmma_bf16(a1, b1, acc[1][1]);

        __syncthreads();   // all reads of bsh[cur] done before TDM reuses it
        cur ^= 1;
    }

#pragma unroll
    for (int mt = 0; mt < 2; ++mt)
#pragma unroll
        for (int nt = 0; nt < 2; ++nt)
#pragma unroll
            for (int r = 0; r < 8; ++r) {
                const int row = m0 + mt * 16 + r + 8 * half;
                const int col = n0blk + wn * 32 + nt * 16 + lr;
                float v = acc[mt][nt][r] + bias[col];
                if (GELU) v = 0.5f * v * (1.0f + erff(v * 0.70710678118654752f));
                if (RES)  v += res[(size_t)row * N + col];
                if (OUTBF16) Cb[(size_t)row * N + col] = (bf16)v;
                else         Cf[(size_t)row * N + col] = v;
            }
}

// ---------------------------------------------------------------------------
// V transpose: qkv[b,s,2D+h*64+hd] (bf16) -> vt[(b*H+h), hd, s] (bf16)
// ---------------------------------------------------------------------------
__global__ __launch_bounds__(256)
void vtrans_kernel(const bf16* __restrict__ qkv, bf16* __restrict__ vt) {
    __shared__ __align__(4) bf16 t[32][34];
    const int tx = threadIdx.x, ty = threadIdx.y;
    const int bh = blockIdx.z;
    const int bb = bh / H_, hh = bh % H_;
    const bf16* in = qkv + (size_t)bb * S_ * 3 * D_ + 2 * D_ + hh * HD_;
    bf16* out = vt + (size_t)bh * HD_ * S_;
    const int s0 = blockIdx.x * 32;
    const int hd0 = blockIdx.y * 32;
#pragma unroll
    for (int i = 0; i < 4; ++i)
        t[ty + i * 8][tx] = in[(size_t)(s0 + ty + i * 8) * (3 * D_) + hd0 + tx];
    __syncthreads();
#pragma unroll
    for (int i = 0; i < 4; ++i)
        out[(size_t)(hd0 + ty + i * 8) * S_ + s0 + tx] = t[tx][ty + i * 8];
}

// ---------------------------------------------------------------------------
// Flash attention, causal. One wave per 16-query tile of one (b,h).
// ---------------------------------------------------------------------------
__global__ __launch_bounds__(32 * AW)
void attn_kernel(const bf16* __restrict__ qkv, const bf16* __restrict__ vt,
                 bf16* __restrict__ attn) {
    __shared__ __align__(16) bf16 psh[AW][16 * 32];
    const int wave = threadIdx.x >> 5;
    const int lane = threadIdx.x & 31;
    const int half = lane >> 4;
    const int lr   = lane & 15;
    const int bh = blockIdx.y;
    const int bb = bh / H_, hh = bh % H_;
    const int q0 = (blockIdx.x * AW + wave) * 16;

    const bf16* qrow  = qkv + ((size_t)bb * S_ + q0 + lr) * (3 * D_) + hh * HD_;
    const bf16* kbase = qkv + (size_t)bb * S_ * (3 * D_) + D_ + hh * HD_;
    const bf16* vtb   = vt + (size_t)bh * HD_ * S_;
    bf16* pw = psh[wave];

    const v16bf qf0 = load_a_frag(qrow, 0, half);
    const v16bf qf1 = load_a_frag(qrow, 32, half);

    v8f o[4];
#pragma unroll
    for (int nt = 0; nt < 4; ++nt) o[nt] = zero_v8f();
    float mrow[8], lrow[8];
#pragma unroll
    for (int r = 0; r < 8; ++r) { mrow[r] = -3.0e38f; lrow[r] = 0.0f; }

    for (int j0 = 0; j0 < q0 + 16; j0 += 32) {
        const bf16* krow0 = kbase + (size_t)(j0 + lr) * (3 * D_);
        const bf16* krow1 = kbase + (size_t)(j0 + 16 + lr) * (3 * D_);
        v8f s0 = zero_v8f(), s1 = zero_v8f();
        s0 = wmma_bf16(qf0, load_b_frag(krow0, 0, half), s0);
        s0 = wmma_bf16(qf1, load_b_frag(krow0, 32, half), s0);
        s1 = wmma_bf16(qf0, load_b_frag(krow1, 0, half), s1);
        s1 = wmma_bf16(qf1, load_b_frag(krow1, 32, half), s1);

        float sv0[8], sv1[8];
#pragma unroll
        for (int r = 0; r < 8; ++r) {
            const int qq = q0 + r + 8 * half;
            float a = s0[r] * 0.125f;                   // 1/sqrt(64)
            float c = s1[r] * 0.125f;
            if (j0 + lr > qq)      a = -1.0e30f;        // causal mask
            if (j0 + 16 + lr > qq) c = -1.0e30f;
            sv0[r] = a; sv1[r] = c;
        }

#pragma unroll
        for (int r = 0; r < 8; ++r) {
            float mx = fmaxf(sv0[r], sv1[r]);
#pragma unroll
            for (int off = 1; off < 16; off <<= 1)
                mx = fmaxf(mx, __shfl_xor(mx, off, 32));
            const float mnew = fmaxf(mrow[r], mx);
            const float corr = __expf(mrow[r] - mnew);
            const float p0 = __expf(sv0[r] - mnew);
            const float p1 = __expf(sv1[r] - mnew);
            float rs = p0 + p1;
#pragma unroll
            for (int off = 1; off < 16; off <<= 1)
                rs += __shfl_xor(rs, off, 32);
            lrow[r] = lrow[r] * corr + rs;
            mrow[r] = mnew;
#pragma unroll
            for (int nt = 0; nt < 4; ++nt) o[nt][r] *= corr;
            pw[(r + 8 * half) * 32 + lr]      = (bf16)p0;
            pw[(r + 8 * half) * 32 + 16 + lr] = (bf16)p1;
        }

        const v16bf pa = load_a_frag(pw + lr * 32, 0, half);
#pragma unroll
        for (int nt = 0; nt < 4; ++nt) {
            const bf16* vrow = vtb + (size_t)(nt * 16 + lr) * S_;
            o[nt] = wmma_bf16(pa, load_b_frag(vrow, j0, half), o[nt]);
        }
    }

    bf16* outb = attn + ((size_t)bb * S_ + q0) * D_ + hh * HD_;
#pragma unroll
    for (int r = 0; r < 8; ++r) {
        const float inv = 1.0f / lrow[r];
#pragma unroll
        for (int nt = 0; nt < 4; ++nt)
            outb[(size_t)(r + 8 * half) * D_ + nt * 16 + lr] =
                (bf16)(o[nt][r] * inv);
    }
}

// ---------------------------------------------------------------------------
// Launch pipeline
// ---------------------------------------------------------------------------
extern "C" void kernel_launch(void* const* d_in, const int* in_sizes, int n_in,
                              void* d_out, int out_size, void* d_ws, size_t ws_size,
                              hipStream_t stream) {
    (void)in_sizes; (void)n_in; (void)out_size; (void)ws_size;
    const float* x     = (const float*)d_in[0];
    // d_in[1] = attention_mask (causal, applied analytically)
    const float* ln1_g = (const float*)d_in[2];
    const float* ln1_b = (const float*)d_in[3];
    const float* w_qkv = (const float*)d_in[4];
    const float* b_qkv = (const float*)d_in[5];
    const float* w_ao  = (const float*)d_in[6];
    const float* b_ao  = (const float*)d_in[7];
    const float* ln2_g = (const float*)d_in[8];
    const float* ln2_b = (const float*)d_in[9];
    const float* w_fc  = (const float*)d_in[10];
    const float* b_fc  = (const float*)d_in[11];
    const float* w_out = (const float*)d_in[12];
    const float* b_out = (const float*)d_in[13];
    float* out = (float*)d_out;

    const size_t BS = (size_t)B_ * S_;
    char* p = (char*)d_ws;
    auto take = [&](size_t bytes) -> char* {
        char* r = p; p += (bytes + 255) & ~(size_t)255; return r;
    };
    bf16* wqkvT = (bf16*)take((size_t)3 * D_ * D_ * 2);  // [3D x D]
    bf16* waoT  = (bf16*)take((size_t)D_ * D_ * 2);      // [D x D]
    bf16* wfcT  = (bf16*)take((size_t)4 * D_ * D_ * 2);  // [4D x D]
    bf16* woutT = (bf16*)take((size_t)4 * D_ * D_ * 2);  // [D x 4D]
    bf16* hbuf  = (bf16*)take(BS * D_ * 2);
    bf16* qkvb  = (bf16*)take(BS * 3 * D_ * 2);
    bf16* vtb   = (bf16*)take(BS * D_ * 2);
    bf16* attnb = (bf16*)take(BS * D_ * 2);
    float* xmid = (float*)take(BS * D_ * 4);
    bf16* h2buf = (bf16*)take(BS * D_ * 2);
    bf16* mlp1  = (bf16*)take(BS * 4 * D_ * 2);

    const dim3 tb(32, 8, 1);
    wconv_kernel<<<dim3(3 * D_ / 32, D_ / 32), tb, 0, stream>>>(w_qkv, wqkvT, D_, 3 * D_);
    wconv_kernel<<<dim3(D_ / 32, D_ / 32), tb, 0, stream>>>(w_ao, waoT, D_, D_);
    wconv_kernel<<<dim3(4 * D_ / 32, D_ / 32), tb, 0, stream>>>(w_fc, wfcT, D_, 4 * D_);
    wconv_kernel<<<dim3(D_ / 32, 4 * D_ / 32), tb, 0, stream>>>(w_out, woutT, 4 * D_, D_);

    ln_kernel<<<(int)BS, 256, 0, stream>>>(x, ln1_g, ln1_b, hbuf);

    gemm_kernel<0, 1, 0><<<dim3(3 * D_ / 64, (int)(BS / 128)), 256, 0, stream>>>(
        hbuf, D_, wqkvT, D_, b_qkv, nullptr, nullptr, qkvb, (int)BS, 3 * D_, D_);

    vtrans_kernel<<<dim3(S_ / 32, HD_ / 32, B_ * H_), tb, 0, stream>>>(qkvb, vtb);

    attn_kernel<<<dim3(S_ / (16 * AW), B_ * H_), 32 * AW, 0, stream>>>(qkvb, vtb, attnb);

    gemm_kernel<0, 0, 1><<<dim3(D_ / 64, (int)(BS / 128)), 256, 0, stream>>>(
        attnb, D_, waoT, D_, b_ao, x, xmid, nullptr, (int)BS, D_, D_);

    ln_kernel<<<(int)BS, 256, 0, stream>>>(xmid, ln2_g, ln2_b, h2buf);

    gemm_kernel<1, 1, 0><<<dim3(4 * D_ / 64, (int)(BS / 128)), 256, 0, stream>>>(
        h2buf, D_, wfcT, D_, b_fc, nullptr, nullptr, mlp1, (int)BS, 4 * D_, D_);

    gemm_kernel<0, 0, 1><<<dim3(D_ / 64, (int)(BS / 128)), 256, 0, stream>>>(
        mlp1, 4 * D_, woutT, 4 * D_, b_out, xmid, out, nullptr, (int)BS, D_, 4 * D_);
}
